// NeuralODE_17635135717704
// MI455X (gfx1250) — compile-verified, matrix operand order checked by reading, not run on hardware
//
#include <hip/hip_runtime.h>
#include <hip/hip_bf16.h>
#include <math.h>

typedef __attribute__((ext_vector_type(16))) __bf16    v16bf;
typedef __attribute__((ext_vector_type(8)))  float     v8f;
typedef __attribute__((ext_vector_type(4)))  unsigned  u32x4;
typedef __attribute__((ext_vector_type(8)))  int       i32x8;
typedef __attribute__((ext_vector_type(4)))  int       i32x4;

#define B_  256
#define S_  64
#define D_  512
#define H_  2048

#if defined(__has_builtin)
#if __has_builtin(__builtin_amdgcn_tensor_load_to_lds) && __has_builtin(__builtin_amdgcn_s_wait_tensorcnt)
#define HAS_TDM 1
#endif
#endif
#ifndef HAS_TDM
#define HAS_TDM 0
#endif

__device__ __forceinline__ unsigned short f2bf_rne(float x) {
    unsigned int u = __float_as_uint(x);
    u += 0x7FFFu + ((u >> 16) & 1u);
    return (unsigned short)(u >> 16);
}

#if HAS_TDM
// Issue one 2D TDM tile load: tile_h rows x tile_w elems (bf16) from a row-major
// tensor with row stride `stride_elems`, into LDS at byte address lds_addr.
// D# bit packing per CDNA5 ISA 8.3/8.4 (group0 128b, group1 256b).
__device__ __forceinline__ void tdm_load_2d(unsigned lds_addr, const void* gptr,
                                            unsigned tile_w, unsigned tile_h,
                                            unsigned stride_elems,
                                            unsigned tensor_w, unsigned tensor_h)
{
    unsigned long long ga = (unsigned long long)(size_t)gptr;
    unsigned long long s0 = (unsigned long long)stride_elems;       // 48-bit
    u32x4 g0;
    g0[0] = 0x1u;                                                   // count=1, user mode
    g0[1] = lds_addr;                                               // LDS byte address
    g0[2] = (unsigned)ga;                                           // global_addr[31:0]
    g0[3] = (unsigned)((ga >> 32) & 0x01FFFFFFu) | (2u << 30);      // addr[56:32] | type=2
    i32x8 g1;
    g1[0] = (int)(1u << 16);                                        // data_size=1 (2 bytes)
    g1[1] = (int)((tensor_w & 0xFFFFu) << 16);                      // tensor_dim0[15:0]
    g1[2] = (int)(((tensor_w >> 16) & 0xFFFFu) | ((tensor_h & 0xFFFFu) << 16));
    g1[3] = (int)(((tensor_h >> 16) & 0xFFFFu) | ((tile_w & 0xFFFFu) << 16));
    g1[4] = (int)(tile_h & 0xFFFFu);                                // tile_dim1, tile_dim2=0
    g1[5] = (int)(unsigned)(s0 & 0xFFFFFFFFu);                      // dim0_stride[31:0]
    g1[6] = (int)(unsigned)((s0 >> 32) & 0xFFFFu);                  // dim0_stride[47:32]
    g1[7] = 0;
    i32x4 gz = {0, 0, 0, 0};
#if __clang_major__ >= 23
    i32x8 gz8 = {0, 0, 0, 0, 0, 0, 0, 0};
    __builtin_amdgcn_tensor_load_to_lds(g0, g1, gz, gz, gz8, 0);
#else
    __builtin_amdgcn_tensor_load_to_lds(g0, g1, gz, gz, 0);
#endif
}
#endif

// ---------------- weight conversion: f32 -> bf16 bits ----------------
__global__ void k_cvt_bf16(const float* __restrict__ src,
                           unsigned short* __restrict__ dst, int n) {
    int i = blockIdx.x * blockDim.x + threadIdx.x;
    if (i < n) dst[i] = f2bf_rne(src[i]);
}

// ---------------- init: y = x[:,0,:]; ybf = bf16(y); out[:,0,:] = y ----------------
__global__ void k_init(const float* __restrict__ x, float* __restrict__ y,
                       unsigned short* __restrict__ ybf, float* __restrict__ out, int n) {
    int i = blockIdx.x * blockDim.x + threadIdx.x;
    if (i < n) {
        int b = i / D_, d = i - b * D_;
        float v = x[(size_t)b * S_ * D_ + d];       // s = 0
        y[i] = v;
        ybf[i] = f2bf_rne(v);
        out[(size_t)b * S_ * D_ + d] = v;
    }
}

// ---------------- zbf = bf16(y + c*k) ----------------
__global__ void k_axpy_bf(const float* __restrict__ y, const float* __restrict__ k,
                          float c, unsigned short* __restrict__ zbf, int n) {
    int i = blockIdx.x * blockDim.x + threadIdx.x;
    if (i < n) zbf[i] = f2bf_rne(fmaf(c, k[i], y[i]));
}

// ------- y += dt/6*(k1+2k2+2k3+k4); ybf = bf16(y); out[:,s,:] = y -------
__global__ void k_combine(float* __restrict__ y, unsigned short* __restrict__ ybf,
                          const float* __restrict__ k1, const float* __restrict__ k2,
                          const float* __restrict__ k3, const float* __restrict__ k4,
                          float dt6, float* __restrict__ out, int sidx, int n) {
    int i = blockIdx.x * blockDim.x + threadIdx.x;
    if (i < n) {
        float v = y[i] + dt6 * (k1[i] + 2.f * k2[i] + 2.f * k3[i] + k4[i]);
        y[i] = v;
        ybf[i] = f2bf_rne(v);
        int b = i / D_, d = i - b * D_;
        out[((size_t)b * S_ + sidx) * D_ + d] = v;
    }
}

// ---------------- GEMM: C[M][N] = act( A[M][K] * W[N][K]^T + bias[N] ) ----------------
// A: bf16 [M][K]. W: bf16 [N][K]. Workgroup = 256 thr (8 waves),
// WG tile 128(M)x64(N); wave tile 32x32 -> 4 v_wmma per K-chunk of 32.
// LDS double-buffered; tiles staged by the Tensor Data Mover (wave 0 issues,
// S_WAIT_TENSORCNT + barrier hands off), fallback = b128 copies.
template<int TANH, int OUT_BF16>
__global__ __launch_bounds__(256)
void k_gemm(const unsigned short* __restrict__ A,
            const unsigned short* __restrict__ W,
            const float* __restrict__ bias,
            void* __restrict__ Cv,
            int N, int K)
{
    __shared__ unsigned short As[2][128][32];   // 16 KB
    __shared__ unsigned short Bs[2][64][32];    //  8 KB

    const int tid   = threadIdx.x;
    const int lane  = tid & 31;
    const int wave  = tid >> 5;
    const int waveM = wave >> 1;            // 0..3 : 32-row slab
    const int waveN = wave & 1;             // 0..1 : 32-col slab
    const int halfk = lane >> 4;            // K-half per ISA 16-bit layout
    const int row16 = lane & 15;

    const int n0 = blockIdx.x * 64;
    const int m0 = blockIdx.y * 128;

    v8f c00 = {}, c01 = {}, c10 = {}, c11 = {};

    auto stage = [&](int buf, int k0) {
#if HAS_TDM
        if (wave == 0) {
            tdm_load_2d((unsigned)(size_t)(void*)&As[buf][0][0],
                        A + (size_t)m0 * K + k0, 32u, 128u, (unsigned)K,
                        (unsigned)K, 1u << 20);
            tdm_load_2d((unsigned)(size_t)(void*)&Bs[buf][0][0],
                        W + (size_t)n0 * K + k0, 32u, 64u, (unsigned)K,
                        (unsigned)K, 1u << 20);
        }
#else
        #pragma unroll
        for (int c = tid; c < 512; c += 256) {          // A: 128x32 bf16 = 512 x b128
            int r = c >> 2, q = c & 3;
            *(uint4*)&As[buf][r][q * 8] =
                *(const uint4*)(A + (size_t)(m0 + r) * K + k0 + q * 8);
        }
        {                                                // B: 64x32 bf16 = 256 x b128
            int r = tid >> 2, q = tid & 3;
            *(uint4*)&Bs[buf][r][q * 8] =
                *(const uint4*)(W + (size_t)(n0 + r) * K + k0 + q * 8);
        }
#endif
    };
    auto stage_wait = [&]() {
#if HAS_TDM
        if (wave == 0) __builtin_amdgcn_s_wait_tensorcnt(0);
#endif
        __syncthreads();
    };

    stage(0, 0);
    stage_wait();

    int buf = 0;
    for (int k0 = 0; k0 < K; k0 += 32) {
        if (k0 + 32 < K) stage(buf ^ 1, k0 + 32);

        union { uint4 u[2]; v16bf v; } a0, a1, b0, b1;
        const unsigned short* ap0 = &As[buf][waveM * 32 + row16][halfk * 8];
        a0.u[0] = *(const uint4*)ap0;       a0.u[1] = *(const uint4*)(ap0 + 16);
        const unsigned short* ap1 = &As[buf][waveM * 32 + 16 + row16][halfk * 8];
        a1.u[0] = *(const uint4*)ap1;       a1.u[1] = *(const uint4*)(ap1 + 16);
        const unsigned short* bp0 = &Bs[buf][waveN * 32 + row16][halfk * 8];
        b0.u[0] = *(const uint4*)bp0;       b0.u[1] = *(const uint4*)(bp0 + 16);
        const unsigned short* bp1 = &Bs[buf][waveN * 32 + 16 + row16][halfk * 8];
        b1.u[0] = *(const uint4*)bp1;       b1.u[1] = *(const uint4*)(bp1 + 16);

        c00 = __builtin_amdgcn_wmma_f32_16x16x32_bf16(false, a0.v, false, b0.v, (short)0, c00, false, false);
        c01 = __builtin_amdgcn_wmma_f32_16x16x32_bf16(false, a0.v, false, b1.v, (short)0, c01, false, false);
        c10 = __builtin_amdgcn_wmma_f32_16x16x32_bf16(false, a1.v, false, b0.v, (short)0, c10, false, false);
        c11 = __builtin_amdgcn_wmma_f32_16x16x32_bf16(false, a1.v, false, b1.v, (short)0, c11, false, false);

        stage_wait();
        buf ^= 1;
    }

    // Epilogue: C/D layout -> VGPR r holds M = r + 8*(lane>>4), N = lane&15
    const int mA   = m0 + waveM * 32 + halfk * 8;       // rows for c00/c01; +16 for c10/c11
    const int nB0  = n0 + waveN * 32 + row16;
    const int nB1  = nB0 + 16;
    const float bb0 = bias[nB0];
    const float bb1 = bias[nB1];
    #pragma unroll
    for (int r = 0; r < 8; ++r) {
        float v00 = c00[r] + bb0, v01 = c01[r] + bb1;
        float v10 = c10[r] + bb0, v11 = c11[r] + bb1;
        if (TANH) { v00 = tanhf(v00); v01 = tanhf(v01); v10 = tanhf(v10); v11 = tanhf(v11); }
        if (OUT_BF16) {
            unsigned short* Cb = (unsigned short*)Cv;
            Cb[(size_t)(mA + r) * N + nB0]      = f2bf_rne(v00);
            Cb[(size_t)(mA + r) * N + nB1]      = f2bf_rne(v01);
            Cb[(size_t)(mA + 16 + r) * N + nB0] = f2bf_rne(v10);
            Cb[(size_t)(mA + 16 + r) * N + nB1] = f2bf_rne(v11);
        } else {
            float* Cf = (float*)Cv;
            Cf[(size_t)(mA + r) * N + nB0]      = v00;
            Cf[(size_t)(mA + r) * N + nB1]      = v01;
            Cf[(size_t)(mA + 16 + r) * N + nB0] = v10;
            Cf[(size_t)(mA + 16 + r) * N + nB1] = v11;
        }
    }
}

// =====================================================================

extern "C" void kernel_launch(void* const* d_in, const int* in_sizes, int n_in,
                              void* d_out, int out_size, void* d_ws, size_t ws_size,
                              hipStream_t stream) {
    const float* x  = (const float*)d_in[0];
    const float* W1 = (const float*)d_in[1];
    const float* b1 = (const float*)d_in[2];
    const float* W2 = (const float*)d_in[3];
    const float* b2 = (const float*)d_in[4];
    const float* W3 = (const float*)d_in[5];
    const float* b3 = (const float*)d_in[6];
    float* out = (float*)d_out;

    char* ws = (char*)d_ws;
    size_t off = 0;
    auto carve = [&](size_t bytes) -> void* {
        void* p = ws + off;
        off = (off + bytes + 255) & ~(size_t)255;
        return p;
    };
    unsigned short* W1b = (unsigned short*)carve((size_t)H_ * D_ * 2);
    unsigned short* W2b = (unsigned short*)carve((size_t)H_ * H_ * 2);
    unsigned short* W3b = (unsigned short*)carve((size_t)D_ * H_ * 2);
    unsigned short* ybf = (unsigned short*)carve((size_t)B_ * D_ * 2);
    unsigned short* zbf = (unsigned short*)carve((size_t)B_ * D_ * 2);
    unsigned short* h1b = (unsigned short*)carve((size_t)B_ * H_ * 2);
    unsigned short* h2b = (unsigned short*)carve((size_t)B_ * H_ * 2);
    float* y  = (float*)carve((size_t)B_ * D_ * 4);
    float* k1 = (float*)carve((size_t)B_ * D_ * 4);
    float* k2 = (float*)carve((size_t)B_ * D_ * 4);
    float* k3 = (float*)carve((size_t)B_ * D_ * 4);
    float* k4 = (float*)carve((size_t)B_ * D_ * 4);

    const int nBD = B_ * D_;
    const int nW1 = H_ * D_, nW2 = H_ * H_, nW3 = D_ * H_;

    k_cvt_bf16<<<(nW1 + 255) / 256, 256, 0, stream>>>(W1, W1b, nW1);
    k_cvt_bf16<<<(nW2 + 255) / 256, 256, 0, stream>>>(W2, W2b, nW2);
    k_cvt_bf16<<<(nW3 + 255) / 256, 256, 0, stream>>>(W3, W3b, nW3);

    k_init<<<(nBD + 255) / 256, 256, 0, stream>>>(x, y, ybf, out, nBD);

    const float dt = 1.0f / (float)(S_ - 1);   // span/steps, steps==1

    auto mlp = [&](const unsigned short* inbf, float* kout) {
        k_gemm<1, 1><<<dim3(H_ / 64, B_ / 128), 256, 0, stream>>>(inbf, W1b, b1, h1b, H_, D_);
        k_gemm<1, 1><<<dim3(H_ / 64, B_ / 128), 256, 0, stream>>>(h1b, W2b, b2, h2b, H_, H_);
        k_gemm<0, 0><<<dim3(D_ / 64, B_ / 128), 256, 0, stream>>>(h2b, W3b, b3, kout, D_, H_);
    };

    const int ew_grid = (nBD + 255) / 256;
    for (int s = 0; s < S_ - 1; ++s) {
        mlp(ybf, k1);
        k_axpy_bf<<<ew_grid, 256, 0, stream>>>(y, k1, 0.5f * dt, zbf, nBD);
        mlp(zbf, k2);
        k_axpy_bf<<<ew_grid, 256, 0, stream>>>(y, k2, 0.5f * dt, zbf, nBD);
        mlp(zbf, k3);
        k_axpy_bf<<<ew_grid, 256, 0, stream>>>(y, k3, dt, zbf, nBD);
        mlp(zbf, k4);
        k_combine<<<ew_grid, 256, 0, stream>>>(y, ybf, k1, k2, k3, k4, dt / 6.0f,
                                               out, s + 1, nBD);
    }
}
